// Attention_81750407512209
// MI455X (gfx1250) — matrix operand order repaired and checked by reading, not compiled, not run on hardware
//
#include <hip/hip_runtime.h>
#include <math.h>

// ---------------------------------------------------------------------------
// Attention (B=8, N=4096, D=128), fp32 in/out, f16 WMMA compute w/ f32 accum.
// Flash-attention streaming softmax; K/V tiles staged in LDS (TDM if
// available, cooperative loads otherwise), double-buffered per block.
// ---------------------------------------------------------------------------

#define EMB   128
#define BATCH 8
#define NSEQ  4096
#define NROWS (BATCH * NSEQ)          // 32768 flat rows

// Padded LDS row strides (halfs). Chosen so 16 lanes reading b128 chunks hit
// bank starts spaced exactly 4 DWORDs apart -> conflict-free.
#define KROW 136                      // 128 + 8  (68 DWORDs)
#define VROW 40                       // 32 + 8   (20 DWORDs)

typedef _Float16 half16  __attribute__((ext_vector_type(16)));
typedef _Float16 half8   __attribute__((ext_vector_type(8)));
typedef float    float8  __attribute__((ext_vector_type(8)));
typedef float    float4v __attribute__((ext_vector_type(4)));
typedef unsigned int uint4v __attribute__((ext_vector_type(4)));
typedef int      int4v  __attribute__((ext_vector_type(4)));
typedef int      int8v  __attribute__((ext_vector_type(8)));

#if defined(__HIP_DEVICE_COMPILE__) && defined(__gfx1250__) && \
    __has_builtin(__builtin_amdgcn_tensor_load_to_lds) &&      \
    __has_builtin(__builtin_amdgcn_s_wait_tensorcnt)
#define USE_TDM 1
#else
#define USE_TDM 0
#endif

__device__ __forceinline__ float8 wmma_f16(half16 a, half16 b, float8 c) {
  // D = A(16x32 f16) * B(32x16 f16) + C(16x16 f32)
  return __builtin_amdgcn_wmma_f32_16x16x32_f16(
      false, a, false, b, (short)0, c, false, false);
}

// B-fragment: two 8-half chunks at p and p+16. lane<16 -> K=0..7,16..23, lane>=16 -> K=8..15,24..31
// (caller bakes the +8 K offset for hi lanes into `p`).
__device__ __forceinline__ half16 load_b_frag(const _Float16* p) {
  union { half16 v; half8 h[2]; } u;
  u.h[0] = *(const half8*)(p);
  u.h[1] = *(const half8*)(p + 16);
  return u.v;
}

#if USE_TDM
// Build a 2D D# (CDNA5 ISA) and issue the DMA.
// data_size=2B, pad_enable=1 (LDS row padding for bank-conflict-free reads).
__device__ __forceinline__ void
tdm_issue_2d(unsigned lds_off, const _Float16* gptr,
             unsigned tensor_d0, unsigned tensor_d1,
             unsigned tile_d0, unsigned tile_d1, unsigned stride0,
             unsigned pad_interval_code, unsigned pad_amount_code) {
  unsigned long long ga = (unsigned long long)(uintptr_t)gptr;
  uint4v g0;
  g0[0] = 1u;                                   // count=1, user descriptor
  g0[1] = lds_off;                              // LDS byte address
  g0[2] = (unsigned)ga;                         // global_addr[31:0]
  g0[3] = (unsigned)((ga >> 32) & 0x01FFFFFFu)  // global_addr[56:32]
          | (2u << 30);                         // type = 2 ("image")
  int8v g1;
  g1[0] = (int)((1u << 16)                      // data_size = 2 bytes
                | (1u << 20)                    // pad_enable
                | (pad_interval_code << 22)     // 2^(c+1) DWORDs per pad
                | (pad_amount_code << 25));     // (c+1) DWORDs of pad
  g1[1] = (int)((tensor_d0 & 0xFFFFu) << 16);   // tensor_dim0[15:0]
  g1[2] = (int)(((tensor_d0 >> 16) & 0xFFFFu) | ((tensor_d1 & 0xFFFFu) << 16));
  g1[3] = (int)(((tensor_d1 >> 16) & 0xFFFFu) | ((tile_d0 & 0xFFFFu) << 16));
  g1[4] = (int)(tile_d1 & 0xFFFFu);             // tile_dim1 (tile_dim2 = 0)
  g1[5] = (int)stride0;                         // tensor_dim0_stride[31:0]
  g1[6] = 0;
  g1[7] = 0;
  int4v z4 = {};
  int8v z8 = {};
  // 6-arg form (this toolchain): (g0, g1, g2, g3, g4, cpol); groups 2/3
  // unused for 2D tiles (tile_dim2 == 0), pass zeros.
  __builtin_amdgcn_tensor_load_to_lds(g0, g1, z4, z4, z8, 0);
}
#endif

// Stage one 32-key K tile (32x128) and V tile (128x32) into padded LDS.
__device__ __forceinline__ void
stage_kv(int wave, int tid, const _Float16* Kg, const _Float16* Vg,
         _Float16* kdst, _Float16* vdst, int kb) {
#if USE_TDM
  if (wave == 0) {
    // K rows kb..kb+31 of [4096][128]: tile 128(w) x 32(h), stride 128.
    // pad: every 64 DWORDs (code 5) add 4 DWORDs (code 3) -> row = 136 halfs.
    tdm_issue_2d((unsigned)(uintptr_t)kdst, Kg + (size_t)kb * EMB,
                 EMB, NSEQ, EMB, 32, EMB, 5u, 3u);
    // V cols kb..kb+31 of [128][4096]: tile 32(w) x 128(h), stride 4096.
    // pad: every 16 DWORDs (code 3) add 4 DWORDs (code 3) -> row = 40 halfs.
    tdm_issue_2d((unsigned)(uintptr_t)vdst, Vg + kb,
                 NSEQ, EMB, 32, EMB, NSEQ, 3u, 3u);
  }
#else
  {   // K: 32 rows x 8 segments of 16 halfs; 256 threads, one segment each
    int row = tid >> 3, seg = tid & 7;
    const half8* src = (const half8*)(Kg + (size_t)(kb + row) * EMB + seg * 16);
    *(half8*)(kdst + row * KROW + seg * 16)     = src[0];
    *(half8*)(kdst + row * KROW + seg * 16 + 8) = src[1];
  }
  {   // V: 128 rows x 2 segments of 16 halfs
    int row = tid >> 1, seg = tid & 1;
    const half8* src = (const half8*)(Vg + (size_t)row * NSEQ + kb + seg * 16);
    *(half8*)(vdst + row * VROW + seg * 16)     = src[0];
    *(half8*)(vdst + row * VROW + seg * 16 + 8) = src[1];
  }
#endif
}

__device__ __forceinline__ void wait_kv(int wave) {
#if USE_TDM
  if (wave == 0) __builtin_amdgcn_s_wait_tensorcnt(0);
#endif
  __syncthreads();
}

// ---------------------------------------------------------------------------
// Kernel 0: Wt[m][e][d] = f16( W[m][d][e] ), with 1/sqrt(D) folded into Wq.
// ---------------------------------------------------------------------------
__global__ void wt_kernel(const float* __restrict__ wq,
                          const float* __restrict__ wk,
                          const float* __restrict__ wv,
                          _Float16* __restrict__ wt, float scale_q) {
  int idx = blockIdx.x * blockDim.x + threadIdx.x;   // 0 .. 3*128*128-1
  int m = idx >> 14;
  int d = (idx >> 7) & 127;
  int e = idx & 127;
  const float* W = (m == 0) ? wq : ((m == 1) ? wk : wv);
  float v = W[d * EMB + e];
  if (m == 0) v *= scale_q;
  wt[m * EMB * EMB + e * EMB + d] = (_Float16)v;     // transposed: [e][d]
}

// ---------------------------------------------------------------------------
// Kernel 1: QKV projection. One wave = one 16-row tile of one matrix.
//   Qh,Kh: row-major f16 [row][128].  Vt: transposed f16 [b][d][n].
// ---------------------------------------------------------------------------
__global__ __launch_bounds__(256) void
qkv_kernel(const float* __restrict__ x, const _Float16* __restrict__ wt,
           _Float16* __restrict__ qh, _Float16* __restrict__ kh,
           _Float16* __restrict__ vt) {
  const int wave = threadIdx.x >> 5;
  const int lane = threadIdx.x & 31;
  const int hi   = lane >> 4;
  const int ln   = lane & 15;
  const int mat  = blockIdx.y;         // 0=Q 1=K 2=V
  const int row0 = (blockIdx.x * 8 + wave) * 16;

  half16 a[4];                         // A-frags of x (f32 -> f16)
  {
    const int r = row0 + ln;
    for (int i = 0; i < 4; ++i) {
      const float* p = x + (size_t)r * EMB + i * 32 + hi * 8;
      float4v x0 = *(const float4v*)(p);
      float4v x1 = *(const float4v*)(p + 4);
      float4v x2 = *(const float4v*)(p + 16);
      float4v x3 = *(const float4v*)(p + 20);
      union { half16 v; _Float16 e[16]; } u;
      for (int j = 0; j < 4; ++j) {
        u.e[j]      = (_Float16)x0[j];
        u.e[4 + j]  = (_Float16)x1[j];
        u.e[8 + j]  = (_Float16)x2[j];
        u.e[12 + j] = (_Float16)x3[j];
      }
      a[i] = u.v;
    }
  }

  const _Float16* W = wt + (size_t)mat * EMB * EMB;  // [e][d] transposed
  for (int et = 0; et < 8; ++et) {
    float8 acc = {};
    for (int i = 0; i < 4; ++i) {
      const _Float16* bp = W + (size_t)(et * 16 + ln) * EMB + i * 32 + hi * 8;
      acc = wmma_f16(a[i], load_b_frag(bp), acc);
    }
    if (mat < 2) {
      _Float16* out = (mat == 0) ? qh : kh;
      for (int j = 0; j < 8; ++j) {
        int rr = row0 + j + hi * 8;
        out[(size_t)rr * EMB + et * 16 + ln] = (_Float16)acc[j];
      }
    } else {
      int    nloc = row0 & (NSEQ - 1);
      size_t base = (size_t)(row0 >> 12) * EMB * NSEQ;
      for (int j = 0; j < 8; ++j)
        vt[base + (size_t)(et * 16 + ln) * NSEQ + nloc + j + hi * 8] =
            (_Float16)acc[j];
    }
  }
}

// ---------------------------------------------------------------------------
// Kernel 2: flash attention. 8 waves/block share LDS-staged K/V tiles
// (double-buffered); each wave owns one 16-row query tile, 32 keys/step.
// ---------------------------------------------------------------------------
__global__ __launch_bounds__(256) void
attn_kernel(const _Float16* __restrict__ qh, const _Float16* __restrict__ kh,
            const _Float16* __restrict__ vt, float* __restrict__ out) {
  __shared__ __align__(16) _Float16 kbuf[2][32 * KROW];   // 2 x 8.5 KB
  __shared__ __align__(16) _Float16 vbuf[2][128 * VROW];  // 2 x 10 KB
  __shared__ __align__(16) _Float16 pbuf[8][16 * 32];     // 8 KB

  const int tid  = threadIdx.x;
  const int wave = tid >> 5;
  const int lane = tid & 31;
  const int hi   = lane >> 4;
  const int ln   = lane & 15;
  const int row0 = (blockIdx.x * 8 + wave) * 16;   // flat query row
  const int b    = row0 >> 12;                     // same for whole block

  // Q A-fragments (pre-scaled by 1/sqrt(D) via Wq).
  half16 qa[4];
  {
    const int r = row0 + ln;
    for (int i = 0; i < 4; ++i) {
      const _Float16* p = qh + (size_t)r * EMB + i * 32 + hi * 8;
      union { half16 v; half8 h[2]; } u;
      u.h[0] = *(const half8*)(p);
      u.h[1] = *(const half8*)(p + 16);
      qa[i] = u.v;
    }
  }

  const _Float16* K = kh + (size_t)b * NSEQ * EMB;   // [n][d]
  const _Float16* V = vt + (size_t)b * EMB * NSEQ;   // [d][n]
  _Float16* myp = pbuf[wave];

  float8 o[8];
  for (int dt = 0; dt < 8; ++dt) o[dt] = (float8){};
  float m[8], l[8];
  for (int j = 0; j < 8; ++j) { m[j] = -INFINITY; l[j] = 0.0f; }

  stage_kv(wave, tid, K, V, kbuf[0], vbuf[0], 0);    // prologue: tile 0

  for (int it = 0; it < NSEQ / 32; ++it) {
    const int kb  = it * 32;
    const int cur = it & 1;
    wait_kv(wave);                                   // tile `it` in LDS
    if (it + 1 < NSEQ / 32)                          // start tile `it+1`
      stage_kv(wave, tid, K, V, kbuf[cur ^ 1], vbuf[cur ^ 1], kb + 32);

    const _Float16* kt = kbuf[cur];
    const _Float16* vl = vbuf[cur];

    // ---- S = Q * K^T : two 16x16 tiles ----
    float8 s0 = {}, s1 = {};
    for (int i = 0; i < 4; ++i) {
      const _Float16* kp = kt + (size_t)ln * KROW + i * 32 + hi * 8;
      s0 = wmma_f16(qa[i], load_b_frag(kp), s0);
    }
    for (int i = 0; i < 4; ++i) {
      const _Float16* kp = kt + (size_t)(16 + ln) * KROW + i * 32 + hi * 8;
      s1 = wmma_f16(qa[i], load_b_frag(kp), s1);
    }

    // ---- online softmax over the 32 new columns ----
    float corr[8];
    for (int j = 0; j < 8; ++j) {
      float t = fmaxf(s0[j], s1[j]);
      t = fmaxf(t, __shfl_xor(t, 1));
      t = fmaxf(t, __shfl_xor(t, 2));
      t = fmaxf(t, __shfl_xor(t, 4));
      t = fmaxf(t, __shfl_xor(t, 8));     // row max within 16-lane half
      float mn = fmaxf(m[j], t);
      float c  = __expf(m[j] - mn);       // 0 when m was -inf
      m[j] = mn;
      float p0 = __expf(s0[j] - mn);
      float p1 = __expf(s1[j] - mn);
      float rs = p0 + p1;
      rs += __shfl_xor(rs, 1);
      rs += __shfl_xor(rs, 2);
      rs += __shfl_xor(rs, 4);
      rs += __shfl_xor(rs, 8);            // row sum of new probs
      l[j] = l[j] * c + rs;
      corr[j] = c;
      int rr = j + hi * 8;                // C-frag row for this lane/VGPR
      myp[rr * 32 + ln]      = (_Float16)p0;
      myp[rr * 32 + 16 + ln] = (_Float16)p1;
    }
    __syncthreads();                      // C-layout -> A-layout via LDS

    half16 pa;                            // P as 16x32 A fragment
    {
      union { half16 v; half8 h[2]; } u;
      u.h[0] = *(const half8*)(myp + ln * 32 + hi * 8);
      u.h[1] = *(const half8*)(myp + ln * 32 + 16 + hi * 8);
      pa = u.v;
    }

    // ---- O = O*corr + P * V ----
    for (int dt = 0; dt < 8; ++dt) {
      float8 t = o[dt];
      for (int j = 0; j < 8; ++j) t[j] *= corr[j];
      const _Float16* vp = vl + (size_t)(dt * 16 + ln) * VROW + hi * 8;
      o[dt] = wmma_f16(pa, load_b_frag(vp), t);
    }
    __syncthreads();                      // done with bufs/pbuf this iter
  }

  // ---- normalize and write fp32 output ----
  float inv[8];
  for (int j = 0; j < 8; ++j) inv[j] = 1.0f / l[j];
  for (int dt = 0; dt < 8; ++dt)
    for (int j = 0; j < 8; ++j)
      out[(size_t)(row0 + j + hi * 8) * EMB + dt * 16 + ln] = o[dt][j] * inv[j];
}

// ---------------------------------------------------------------------------
// Host launcher
// ---------------------------------------------------------------------------
extern "C" void kernel_launch(void* const* d_in, const int* in_sizes, int n_in,
                              void* d_out, int out_size, void* d_ws,
                              size_t ws_size, hipStream_t stream) {
  const float* x  = (const float*)d_in[0];
  const float* wq = (const float*)d_in[1];
  const float* wk = (const float*)d_in[2];
  const float* wv = (const float*)d_in[3];
  float* out = (float*)d_out;

  // Workspace carve-up (f16): Wt 96KB | Qh 8MB | Kh 8MB | Vt 8MB  (~24.1 MB)
  char* ws = (char*)d_ws;
  _Float16* wt = (_Float16*)(ws);
  _Float16* qh = (_Float16*)(ws + 98304);
  _Float16* kh = (_Float16*)(ws + 98304 + (size_t)NROWS * EMB * 2);
  _Float16* vt = (_Float16*)(ws + 98304 + (size_t)NROWS * EMB * 4);

  const float scale_q = 0.08838834764831845f;  // 1/sqrt(128)

  wt_kernel<<<(3 * EMB * EMB) / 256, 256, 0, stream>>>(wq, wk, wv, wt, scale_q);
  qkv_kernel<<<dim3(NROWS / 16 / 8, 3), 256, 0, stream>>>(x, wt, qh, kh, vt);
  attn_kernel<<<NROWS / 16 / 8, 256, 0, stream>>>(qh, kh, vt, out);
}
